// SpectralDualPathContextEngine_9234179686580
// MI455X (gfx1250) — matrix operand order repaired and checked by reading, not compiled, not run on hardware
//
#include <hip/hip_runtime.h>
#include <math.h>

typedef _Float16 h16;
typedef __attribute__((ext_vector_type(16))) _Float16 v16h;
typedef __attribute__((ext_vector_type(8)))  _Float16 v8h;
typedef __attribute__((ext_vector_type(8)))  float    v8f;

#define BB   4
#define SS   512
#define DD   2048
#define NHH  8
#define DEGG 4
#define HDD  256
#define AHH  16
#define AHDD 128

// ---------------- reduction helpers ----------------
__device__ __forceinline__ float warp_sum32(float v) {
#pragma unroll
  for (int o = 16; o > 0; o >>= 1) v += __shfl_xor(v, o, 32);
  return v;
}

__device__ __forceinline__ float blk_sum256(float v) {
  __shared__ float sb[8];
  v = warp_sum32(v);
  int w = threadIdx.x >> 5;
  __syncthreads();
  if ((threadIdx.x & 31) == 0) sb[w] = v;
  __syncthreads();
  float s = 0.f;
#pragma unroll
  for (int i = 0; i < 8; i++) s += sb[i];
  return s;
}

__device__ __forceinline__ float sigmoidf_(float x) { return 1.0f / (1.0f + __expf(-x)); }
__device__ __forceinline__ float siluf_(float x) { return x * sigmoidf_(x); }

// A-matrix fragment (16x32 f16): lane half kb=(lane>>4)*8,
// halves 0..7 -> K kb..kb+7, halves 8..15 -> K kb+16..kb+23  (ISA 7.12.2)
__device__ __forceinline__ v16h mk_fragA(const h16* p) {
  v8h lo = *(const v8h*)p;
  v8h hi = *(const v8h*)(p + 16);
  v16h r;
#pragma unroll
  for (int i = 0; i < 8; i++) { r[i] = lo[i]; r[8 + i] = hi[i]; }
  return r;
}

// B-matrix fragment (32x16 f16): lane col = lane&15, kb=(lane>>4)*8, halves 0..7 -> K kb..kb+7, halves 8..15 -> K kb+16..kb+23
__device__ __forceinline__ v16h mk_fragB(const h16* p) {
  v8h lo = *(const v8h*)p;
  v8h hi = *(const v8h*)(p + 16);
  v16h r;
#pragma unroll
  for (int i = 0; i < 8; i++) { r[i] = lo[i]; r[8 + i] = hi[i]; }
  return r;
}

// ---------------- elementwise / small kernels ----------------
__global__ void k_rmsnorm(const float* __restrict__ x, const float* __restrict__ w,
                          float* __restrict__ xn, h16* __restrict__ xnh) {
  int row = blockIdx.x, t = threadIdx.x;
  const float* xr = x + (size_t)row * DD;
  float v[8]; float ss = 0.f;
#pragma unroll
  for (int i = 0; i < 8; i++) { v[i] = xr[t * 8 + i]; ss += v[i] * v[i]; }
  ss = blk_sum256(ss);
  float inv = rsqrtf(ss / (float)DD + 1e-6f);
  float* xo = xn + (size_t)row * DD;
  h16*   xh = xnh + (size_t)row * DD;
#pragma unroll
  for (int i = 0; i < 8; i++) {
    float o = v[i] * inv * w[t * 8 + i];
    xo[t * 8 + i] = o;
    xh[t * 8 + i] = (h16)o;
  }
}

__global__ void k_colmean(const float* __restrict__ xn, float* __restrict__ xm) {
  int d = blockIdx.x * 256 + threadIdx.x;
  int b = blockIdx.y;
  const float* p = xn + (size_t)b * SS * DD + d;
  float s = 0.f;
  for (int i = 0; i < SS; i++) s += p[(size_t)i * DD];
  xm[b * DD + d] = s / (float)SS;
}

__global__ void k_mlp1(const float* __restrict__ xm, const float* __restrict__ wc1,
                       const float* __restrict__ bc1, float* __restrict__ h) {
  int idx = blockIdx.x * 256 + threadIdx.x;   // B*512
  int b = idx >> 9, j = idx & 511;
  const float* xp = xm + b * DD;
  float acc = bc1[j];
  for (int d = 0; d < DD; d++) acc += xp[d] * wc1[(size_t)d * 512 + j];
  h[idx] = siluf_(acc);
}

__global__ void k_mlp2(const float* __restrict__ h, const float* __restrict__ wc2,
                       const float* __restrict__ bc2, float* __restrict__ coeffs) {
  int idx = blockIdx.x * 256 + threadIdx.x;   // B*8192
  int b = idx >> 13, o = idx & 8191;
  const float* hp = h + b * 512;
  float acc = bc2[o];
  for (int j = 0; j < 512; j++) acc += hp[j] * wc2[(size_t)j * 8192 + o];
  int k = (o >> 8) & 3;                        // o = hh*1024 + k*256 + d
  coeffs[idx] = acc * (0.1f / (float)(k + 1));
}

__global__ void k_cheby(const float* __restrict__ xn, const float* __restrict__ coeffs,
                        float* __restrict__ cheb, h16* __restrict__ kvh) {
  int row = blockIdx.x;          // b*S + s
  int b = row >> 9;
  int t = threadIdx.x;
  int hh = t >> 5;               // head == wave id (8 waves, 8 heads)
  int lane = t & 31;
  const float* xr = xn + (size_t)row * DD + hh * HDD + lane * 8;
  float u[8]; float s = 0.f;
#pragma unroll
  for (int i = 0; i < 8; i++) { u[i] = xr[i]; s += u[i]; }
  s = warp_sum32(s);
  float z = tanhf(s / (float)HDD);
  float T0 = 1.f, T1 = z, T2 = 2.f * z * T1 - T0, T3 = 2.f * z * T2 - T1;
  const float* cb = coeffs + (size_t)b * 8192 + hh * 1024 + lane * 8;
  float* co = cheb + (size_t)row * DD + hh * HDD + lane * 8;
  h16*   kv = kvh + (size_t)row * DD + hh * HDD + lane * 8;
#pragma unroll
  for (int i = 0; i < 8; i++) {
    float o = T0 * cb[i] + T1 * cb[256 + i] + T2 * cb[512 + i] + T3 * cb[768 + i];
    co[i] = o;
    kv[i] = (h16)(u[i] + o);
  }
}

__global__ void k_f2h(const float* __restrict__ f, h16* __restrict__ h, int n) {
  int i = blockIdx.x * 256 + threadIdx.x;
  if (i < n) h[i] = (h16)f[i];
}

__global__ void k_h1h2(const float* __restrict__ ch, const float* __restrict__ slr,
                       const float* __restrict__ wnc, const float* __restrict__ wns,
                       h16* __restrict__ hcat) {
  int row = blockIdx.x, t = threadIdx.x;
  const float* c = ch + (size_t)row * DD;
  const float* s = slr + (size_t)row * DD;
  float cv[8], sv[8]; float s1 = 0.f, s2 = 0.f;
#pragma unroll
  for (int i = 0; i < 8; i++) {
    cv[i] = c[t * 8 + i]; s1 += cv[i] * cv[i];
    sv[i] = s[t * 8 + i]; s2 += sv[i] * sv[i];
  }
  s1 = blk_sum256(s1);
  s2 = blk_sum256(s2);
  float i1 = rsqrtf(s1 / (float)DD + 1e-6f);
  float i2 = rsqrtf(s2 / (float)DD + 1e-6f);
  h16* o = hcat + (size_t)row * 2 * DD;
#pragma unroll
  for (int i = 0; i < 8; i++) {
    o[t * 8 + i]        = (h16)(cv[i] * i1 * wnc[t * 8 + i]);
    o[DD + t * 8 + i]   = (h16)(sv[i] * i2 * wns[t * 8 + i]);
  }
}

__global__ void k_gvfuse(const float* __restrict__ gv, const float* __restrict__ wnp,
                         h16* __restrict__ fused) {
  int row = blockIdx.x, t = threadIdx.x;
  const float* g = gv + (size_t)row * 2 * DD;
  const float* v = g + DD;
  float tv[8]; float ss = 0.f;
#pragma unroll
  for (int i = 0; i < 8; i++) {
    float x = siluf_(g[t * 8 + i]) * v[t * 8 + i];
    tv[i] = x; ss += x * x;
  }
  ss = blk_sum256(ss);
  float inv = rsqrtf(ss / (float)DD + 1e-6f);
  h16* o = fused + (size_t)row * DD;
#pragma unroll
  for (int i = 0; i < 8; i++) o[t * 8 + i] = (h16)(tv[i] * inv * wnp[t * 8 + i]);
}

// ---------------- WMMA GEMM: C[M,N] = A[M,K] * B[K,N], f16 in / f32 acc ----------------
// 128x128 block tile, 8 waves (4x2) of 32x64 wave tiles, K step 32, double-buffered LDS.
// MODE 0: store f32   MODE 1: store f16   MODE 2: store f32 + residual
template <int MODE>
__global__ __launch_bounds__(256, 1)
void k_gemm(const h16* __restrict__ A, const h16* __restrict__ Bm,
            float* __restrict__ Cf, h16* __restrict__ Ch,
            const float* __restrict__ R, int N, int K) {
  __shared__ __align__(16) h16 As[2][128 * 40];
  __shared__ __align__(16) h16 BsT[2][128 * 40];
  int t = threadIdx.x;
  int lane = t & 31, wid = t >> 5;
  int wm = wid & 3, wn = wid >> 2;
  int row0 = blockIdx.y * 128, col0 = blockIdx.x * 128;
  int lr  = lane & 15;
  int kbA = (lane >> 4) * 8;
  int kbB = (lane >> 4) * 8;

  // per-thread staging indices (reused every tile)
  int rA0 = t >> 2,          sgA0 = t & 3;          // pass 0
  int rA1 = (256 + t) >> 2,  sgA1 = (256 + t) & 3;  // pass 1
  int krB0 = t >> 4,         sgB0 = t & 15;
  int krB1 = (256 + t) >> 4, sgB1 = (256 + t) & 15;

  v8f zero = {0.f, 0.f, 0.f, 0.f, 0.f, 0.f, 0.f, 0.f};
  v8f acc[2][4];
#pragma unroll
  for (int i = 0; i < 2; i++)
#pragma unroll
    for (int j = 0; j < 4; j++) acc[i][j] = zero;

  // prologue: stage tile kk=0 into buffer 0
  {
    *(uint4*)(As[0] + rA0 * 40 + sgA0 * 8) =
        *(const uint4*)(A + (size_t)(row0 + rA0) * K + sgA0 * 8);
    *(uint4*)(As[0] + rA1 * 40 + sgA1 * 8) =
        *(const uint4*)(A + (size_t)(row0 + rA1) * K + sgA1 * 8);
    v8h b0 = *(const v8h*)(Bm + (size_t)krB0 * N + col0 + sgB0 * 8);
    v8h b1 = *(const v8h*)(Bm + (size_t)krB1 * N + col0 + sgB1 * 8);
#pragma unroll
    for (int e = 0; e < 8; e++) BsT[0][(sgB0 * 8 + e) * 40 + krB0] = b0[e];
#pragma unroll
    for (int e = 0; e < 8; e++) BsT[0][(sgB1 * 8 + e) * 40 + krB1] = b1[e];
  }

  int buf = 0;
  for (int kk = 0; kk < K; kk += 32) {
    __syncthreads();              // tile[buf] ready; previous compute on buf^1 done
    int kn = kk + 32;
    if (kn < K) {                 // prefetch next tile into the other buffer
      int nb = buf ^ 1;
      *(uint4*)(As[nb] + rA0 * 40 + sgA0 * 8) =
          *(const uint4*)(A + (size_t)(row0 + rA0) * K + kn + sgA0 * 8);
      *(uint4*)(As[nb] + rA1 * 40 + sgA1 * 8) =
          *(const uint4*)(A + (size_t)(row0 + rA1) * K + kn + sgA1 * 8);
      v8h b0 = *(const v8h*)(Bm + (size_t)(kn + krB0) * N + col0 + sgB0 * 8);
      v8h b1 = *(const v8h*)(Bm + (size_t)(kn + krB1) * N + col0 + sgB1 * 8);
#pragma unroll
      for (int e = 0; e < 8; e++) BsT[nb][(sgB0 * 8 + e) * 40 + krB0] = b0[e];
#pragma unroll
      for (int e = 0; e < 8; e++) BsT[nb][(sgB1 * 8 + e) * 40 + krB1] = b1[e];
    }

    v16h af[2], bf[4];
#pragma unroll
    for (int mi = 0; mi < 2; mi++)
      af[mi] = mk_fragA(As[buf] + (wm * 32 + mi * 16 + lr) * 40 + kbA);
#pragma unroll
    for (int ni = 0; ni < 4; ni++)
      bf[ni] = mk_fragB(BsT[buf] + (wn * 64 + ni * 16 + lr) * 40 + kbB);
#pragma unroll
    for (int mi = 0; mi < 2; mi++)
#pragma unroll
      for (int ni = 0; ni < 4; ni++)
        acc[mi][ni] = __builtin_amdgcn_wmma_f32_16x16x32_f16(
            false, af[mi], false, bf[ni], (short)0, acc[mi][ni], false, false);
    buf ^= 1;
  }

  int rhalf = (lane >> 4) * 8;   // C layout: row = e + (lane>>4)*8, col = lane&15
#pragma unroll
  for (int mi = 0; mi < 2; mi++)
#pragma unroll
    for (int ni = 0; ni < 4; ni++)
#pragma unroll
      for (int e = 0; e < 8; e++) {
        int r = row0 + wm * 32 + mi * 16 + rhalf + e;
        int cI = col0 + wn * 64 + ni * 16 + lr;
        float vv = acc[mi][ni][e];
        if (MODE == 0)       Cf[(size_t)r * N + cI] = vv;
        else if (MODE == 1)  Ch[(size_t)r * N + cI] = (h16)vv;
        else                 Cf[(size_t)r * N + cI] = R[(size_t)r * N + cI] + vv;
      }
}

// ---------------- flash causal attention (S=512, hd=128, 16 heads) ----------------
// One block per (b, head, q-block of 128). 8 waves x 16 q rows. Key tiles of 64.
__global__ __launch_bounds__(256, 1)
void k_attn(const h16* __restrict__ Q, const h16* __restrict__ Kh,
            const h16* __restrict__ Vh, h16* __restrict__ O) {
  __shared__ __align__(16) h16 Ks[64 * 136];       // [key][hd]
  __shared__ __align__(16) h16 VsT[128 * 72];      // [hd][key]
  __shared__ __align__(16) h16 Ps[8][16 * 72];     // per wave [row][key]
  int blk = blockIdx.x;
  int qb = blk & 3;
  int hh = (blk >> 2) & 15;
  int b  = blk >> 6;
  int t = threadIdx.x, lane = t & 31, w = t >> 5;
  int lr = lane & 15;
  int kbA = (lane >> 4) * 8;
  int kbB = (lane >> 4) * 8;

  // Q fragments resident in registers: 16 rows x 128 hd per wave
  const h16* qrow = Q + (size_t)(b * SS + qb * 128 + w * 16 + lr) * DD + hh * AHDD;
  v16h aq[4];
#pragma unroll
  for (int kc = 0; kc < 4; kc++) aq[kc] = mk_fragA(qrow + kc * 32 + kbA);

  v8f zero = {0.f, 0.f, 0.f, 0.f, 0.f, 0.f, 0.f, 0.f};
  v8f oacc[8];
#pragma unroll
  for (int i = 0; i < 8; i++) oacc[i] = zero;
  float mrow[8], lrow[8];
#pragma unroll
  for (int e = 0; e < 8; e++) { mrow[e] = -1e30f; lrow[e] = 0.f; }

  const float scl0 = 0.088388347648318447f;  // 1/sqrt(128)
  int qbase = qb * 128 + w * 16 + (lane >> 4) * 8;
  int lrb = (lane >> 4) * 8;
  int jend = 2 * qb + 1;

  for (int j = 0; j <= jend; j++) {
    __syncthreads();
    int key0 = j * 64;
#pragma unroll
    for (int p = 0; p < 4; p++) {                        // K tile 64x128
      int c = p * 256 + t;
      int ky = c >> 4, sg = c & 15;
      *(uint4*)(Ks + ky * 136 + sg * 8) =
          *(const uint4*)(Kh + (size_t)(b * SS + key0 + ky) * DD + hh * AHDD + sg * 8);
    }
#pragma unroll
    for (int p = 0; p < 4; p++) {                        // V tile transposed
      int c = p * 256 + t;
      int ky = c >> 4, sg = c & 15;
      v8h vv = *(const v8h*)(Vh + (size_t)(b * SS + key0 + ky) * DD + hh * AHDD + sg * 8);
#pragma unroll
      for (int e = 0; e < 8; e++) VsT[(sg * 8 + e) * 72 + ky] = vv[e];
    }
    __syncthreads();

    // S = Q K^T  (16 x 64 per wave)
    v8f sacc[4];
#pragma unroll
    for (int n = 0; n < 4; n++) sacc[n] = zero;
#pragma unroll
    for (int n = 0; n < 4; n++)
#pragma unroll
      for (int kc = 0; kc < 4; kc++) {
        v16h bf = mk_fragB(Ks + (n * 16 + lr) * 136 + kc * 32 + kbB);
        sacc[n] = __builtin_amdgcn_wmma_f32_16x16x32_f16(
            false, aq[kc], false, bf, (short)0, sacc[n], false, false);
      }

    // causal mask + scale, in place
#pragma unroll
    for (int n = 0; n < 4; n++) {
      int keyg = key0 + n * 16 + lr;
#pragma unroll
      for (int e = 0; e < 8; e++) {
        float s = sacc[n][e] * scl0;
        sacc[n][e] = (keyg <= qbase + e) ? s : -1e30f;
      }
    }
    // online softmax, probabilities written back into sacc
#pragma unroll
    for (int e = 0; e < 8; e++) {
      float mx = sacc[0][e];
#pragma unroll
      for (int n = 1; n < 4; n++) mx = fmaxf(mx, sacc[n][e]);
#pragma unroll
      for (int o = 1; o < 16; o <<= 1) mx = fmaxf(mx, __shfl_xor(mx, o, 16));
      float mnew = fmaxf(mrow[e], mx);
      float rsc = __expf(mrow[e] - mnew);
      float rs = 0.f;
#pragma unroll
      for (int n = 0; n < 4; n++) {
        float p = __expf(sacc[n][e] - mnew);
        sacc[n][e] = p; rs += p;
      }
#pragma unroll
      for (int o = 1; o < 16; o <<= 1) rs += __shfl_xor(rs, o, 16);
      lrow[e] = lrow[e] * rsc + rs;
      mrow[e] = mnew;
#pragma unroll
      for (int tt = 0; tt < 8; tt++) oacc[tt][e] *= rsc;
    }

    // stage P to LDS in A-operand order
#pragma unroll
    for (int e = 0; e < 8; e++) {
      int prow = lrb + e;
#pragma unroll
      for (int n = 0; n < 4; n++)
        Ps[w][prow * 72 + n * 16 + lr] = (h16)sacc[n][e];
    }
    __syncthreads();

    // O += P V
    v16h pa[2];
#pragma unroll
    for (int kc = 0; kc < 2; kc++) pa[kc] = mk_fragA(&Ps[w][lr * 72 + kc * 32 + kbA]);
#pragma unroll
    for (int tt = 0; tt < 8; tt++)
#pragma unroll
      for (int kc = 0; kc < 2; kc++) {
        v16h vb = mk_fragB(VsT + (tt * 16 + lr) * 72 + kc * 32 + kbB);
        oacc[tt] = __builtin_amdgcn_wmma_f32_16x16x32_f16(
            false, pa[kc], false, vb, (short)0, oacc[tt], false, false);
      }
  }

  // normalize + store (layout [b*S + q, head*128 + hd], f16 for wo GEMM)
  int rowg = b * SS + qb * 128 + w * 16 + (lane >> 4) * 8;
#pragma unroll
  for (int e = 0; e < 8; e++) {
    float inv = 1.f / lrow[e];
#pragma unroll
    for (int tt = 0; tt < 8; tt++)
      O[(size_t)(rowg + e) * DD + hh * AHDD + tt * 16 + lr] = (h16)(oacc[tt][e] * inv);
  }
}

// ---------------- launch ----------------
extern "C" void kernel_launch(void* const* d_in, const int* in_sizes, int n_in,
                              void* d_out, int out_size, void* d_ws, size_t ws_size,
                              hipStream_t stream) {
  (void)in_sizes; (void)n_in; (void)out_size; (void)ws_size;
  const float* x    = (const float*)d_in[0];
  const float* w_in = (const float*)d_in[1];
  // d_in[2..5] = wg1,bg1,wg2,bg2 : dead for the output (only feed coeffs_new/momentum)
  const float* wc1  = (const float*)d_in[6];
  const float* bc1  = (const float*)d_in[7];
  const float* wc2  = (const float*)d_in[8];
  const float* bc2  = (const float*)d_in[9];
  const float* wq   = (const float*)d_in[10];
  const float* wk   = (const float*)d_in[11];
  const float* wv   = (const float*)d_in[12];
  const float* wo   = (const float*)d_in[13];
  const float* wnc  = (const float*)d_in[14];
  const float* wns  = (const float*)d_in[15];
  const float* wfgv = (const float*)d_in[16];
  const float* wnp  = (const float*)d_in[17];
  const float* wout = (const float*)d_in[18];

  char* ws = (char*)d_ws;
  size_t off = 0;
  auto alloc = [&](size_t bytes) -> void* {
    void* p = ws + off;
    off += (bytes + 255) & ~(size_t)255;
    return p;
  };
  const size_t ROWS = (size_t)BB * SS;   // 2048

  float* xn    = (float*)alloc(ROWS * DD * 4);
  h16*   xnh   = (h16*)  alloc(ROWS * DD * 2);
  float* cheb  = (float*)alloc(ROWS * DD * 4);
  h16*   kvh   = (h16*)  alloc(ROWS * DD * 2);
  float* xm    = (float*)alloc((size_t)BB * DD * 4);
  float* hbuf  = (float*)alloc((size_t)BB * 512 * 4);
  float* coef  = (float*)alloc((size_t)BB * 8192 * 4);
  h16*   qh    = (h16*)  alloc(ROWS * DD * 2);
  h16*   kh    = (h16*)  alloc(ROWS * DD * 2);
  h16*   vh    = (h16*)  alloc(ROWS * DD * 2);
  h16*   ah    = (h16*)  alloc(ROWS * DD * 2);
  float* slr   = (float*)alloc(ROWS * DD * 4);
  h16*   hcat  = (h16*)  alloc(ROWS * 2 * DD * 2);
  float* gvb   = (float*)alloc(ROWS * 2 * DD * 4);
  h16*   fusedh= (h16*)  alloc(ROWS * DD * 2);
  h16*   wqh   = (h16*)  alloc((size_t)DD * DD * 2);
  h16*   wkh   = (h16*)  alloc((size_t)DD * DD * 2);
  h16*   wvh   = (h16*)  alloc((size_t)DD * DD * 2);
  h16*   woh   = (h16*)  alloc((size_t)DD * DD * 2);
  h16*   wouth = (h16*)  alloc((size_t)DD * DD * 2);
  h16*   wfgvh = (h16*)  alloc((size_t)2 * DD * 2 * DD * 2);

  // weight f32 -> f16
  int nw = DD * DD;
  k_f2h<<<(nw + 255) / 256, 256, 0, stream>>>(wq, wqh, nw);
  k_f2h<<<(nw + 255) / 256, 256, 0, stream>>>(wk, wkh, nw);
  k_f2h<<<(nw + 255) / 256, 256, 0, stream>>>(wv, wvh, nw);
  k_f2h<<<(nw + 255) / 256, 256, 0, stream>>>(wo, woh, nw);
  k_f2h<<<(nw + 255) / 256, 256, 0, stream>>>(wout, wouth, nw);
  int nf = 2 * DD * 2 * DD;
  k_f2h<<<(nf + 255) / 256, 256, 0, stream>>>(wfgv, wfgvh, nf);

  // norms, coeff MLP, chebyshev mix
  k_rmsnorm<<<(int)ROWS, 256, 0, stream>>>(x, w_in, xn, xnh);
  k_colmean<<<dim3(DD / 256, BB), 256, 0, stream>>>(xn, xm);
  k_mlp1<<<(BB * 512) / 256, 256, 0, stream>>>(xm, wc1, bc1, hbuf);
  k_mlp2<<<(BB * 8192) / 256, 256, 0, stream>>>(hbuf, wc2, bc2, coef);
  k_cheby<<<(int)ROWS, 256, 0, stream>>>(xn, coef, cheb, kvh);

  // q/k/v projections (WMMA, f16 out)
  dim3 g16(16, 16);
  k_gemm<1><<<g16, 256, 0, stream>>>(xnh, wqh, nullptr, qh, nullptr, DD, DD);
  k_gemm<1><<<g16, 256, 0, stream>>>(kvh, wkh, nullptr, kh, nullptr, DD, DD);
  k_gemm<1><<<g16, 256, 0, stream>>>(kvh, wvh, nullptr, vh, nullptr, DD, DD);

  // causal attention (nc==1 => plain causal within S=512)
  k_attn<<<BB * AHH * 4, 256, 0, stream>>>(qh, kh, vh, ah);

  // slr = attn @ wo (f32 out)
  k_gemm<0><<<g16, 256, 0, stream>>>(ah, woh, slr, nullptr, nullptr, DD, DD);

  // h1/h2 norms -> concat (f16)
  k_h1h2<<<(int)ROWS, 256, 0, stream>>>(cheb, slr, wnc, wns, hcat);

  // gv = hcat @ w_fgv  (M=2048, N=4096, K=4096)
  k_gemm<0><<<dim3(32, 16), 256, 0, stream>>>(hcat, wfgvh, gvb, nullptr, nullptr, 2 * DD, 2 * DD);

  // fused = rmsnorm(silu(gate)*value) (f16)
  k_gvfuse<<<(int)ROWS, 256, 0, stream>>>(gvb, wnp, fusedh);

  // out = x + fused @ w_out (f32, residual)
  k_gemm<2><<<g16, 256, 0, stream>>>(fusedh, wouth, (float*)d_out, nullptr, x, DD, DD);
}
